// MultimodalCrossAttention_77472620085716
// MI455X (gfx1250) — matrix operand order repaired and checked by reading, not compiled, not run on hardware
//
#include <hip/hip_runtime.h>
#include <hip/hip_bf16.h>

// MI455X / gfx1250 multimodal cross-attention.
// bf16 WMMA for all GEMMs (f32 accumulate), softmax f32 in place on the NxN
// weight outputs, and TDM (tensor_load_to_lds) double-buffered LDS staging.

#define HIDDEN 1024
#define NSEQ   8192

typedef __bf16 bf16;
typedef __attribute__((ext_vector_type(16))) __bf16 v16bf;
typedef __attribute__((ext_vector_type(8)))  float  v8f;
typedef __attribute__((ext_vector_type(4)))  unsigned int u32x4;
typedef __attribute__((ext_vector_type(8)))  int i32x8;
typedef __attribute__((ext_vector_type(4)))  int i32x4;

// ---------------- GEMM tiling ----------------
#define BM 128
#define BN 128
#define BK 32
#define LDS_STRIDE 40   // BK + 8 elem pad -> 80B rows; TDM pad: 16 DW interval + 4 DW
#define GEMM_THREADS 256

#if defined(__HIP_DEVICE_COMPILE__) && __has_builtin(__builtin_amdgcn_tensor_load_to_lds) && __has_builtin(__builtin_amdgcn_s_wait_tensorcnt)
#define GFX1250_TDM 1
#else
#define GFX1250_TDM 0
#endif

__device__ __forceinline__ unsigned lds_byte_addr(const void* p)
{
    // generic pointers into LDS carry the LDS byte offset in the low 32 bits
    return (unsigned)(unsigned long long)p;
}

// One TDM descriptor: load a BM x BK bf16 tile (row stride = K elements) from
// global into LDS with 16B padding after every 64B row -> LDS_STRIDE layout.
__device__ __forceinline__ void tdm_load_tile_bf16(const bf16* gsrc, unsigned lds_byte,
                                                   int row_stride_elems)
{
#if GFX1250_TDM
    unsigned long long ga = (unsigned long long)(const void*)gsrc;
    u32x4 g0;
    g0.x = 1u;                                            // count=1, user descriptor
    g0.y = lds_byte;                                      // lds_addr
    g0.z = (unsigned)ga;                                  // global_addr[31:0]
    g0.w = (unsigned)((ga >> 32) & 0x01ffffffu) | (2u << 30); // addr[56:32] | type=2

    const unsigned w0 = (1u << 16)    // data_size = 2 bytes
                      | (1u << 20)    // pad_enable
                      | (3u << 22)    // pad_interval: 16 DWORDs (=64B)
                      | (3u << 25);   // pad_amount:   4 DWORDs (=16B)
    const unsigned td0 = 0x7fffffffu; // huge tensor dims: tiles never clip
    const unsigned td1 = 0x7fffffffu;
    i32x8 g1;
    g1[0] = (int)w0;
    g1[1] = (int)(td0 << 16);                          // [63:48] tensor_dim0 lo
    g1[2] = (int)((td0 >> 16) | (td1 << 16));          // dim0 hi | dim1 lo
    g1[3] = (int)((td1 >> 16) | ((unsigned)BK << 16)); // dim1 hi | tile_dim0=32
    g1[4] = BM;                                        // tile_dim1=128, tile_dim2=0
    unsigned long long st = (unsigned long long)row_stride_elems;
    g1[5] = (int)(st & 0xffffffffu);                   // tensor_dim0_stride lo
    g1[6] = (int)((st >> 32) & 0xffffu);               // stride hi | dim1_stride lo=0
    g1[7] = 0;

    i32x4 z4 = {0, 0, 0, 0};
    i32x8 z8 = {0, 0, 0, 0, 0, 0, 0, 0};
    // clang-23 / therock-10.0 6-arg form: (g0, g1, g2, g3, extra, cpol)
    __builtin_amdgcn_tensor_load_to_lds(g0, g1, z4, z4, z8, 0);
#else
    (void)gsrc; (void)lds_byte; (void)row_stride_elems;
#endif
}

// C[M,N] = A[M,K] * B^T + bias, where B is stored [N,K] row-major.
//  A_F32 : A source is f32 (converted to bf16 while staging through VGPRs),
//          otherwise bf16 and staged by the TDM.
//  OUT_T : bf16 output stored transposed (Cb[n*M + m]).
template<bool A_F32, bool HAS_BIAS, bool OUT_F32, bool OUT_BF16, bool OUT_T>
__global__ void __launch_bounds__(GEMM_THREADS)
gemm_bf16_wmma(const void* __restrict__ Aptr, const bf16* __restrict__ Bptr,
               const float* __restrict__ bias,
               float* __restrict__ Cf, bf16* __restrict__ Cb,
               int M, int N, int K)
{
    __shared__ __align__(16) bf16 Ash[2][BM * LDS_STRIDE];
    __shared__ __align__(16) bf16 Bsh[2][BN * LDS_STRIDE];

    const int tid  = threadIdx.x;
    const int lane = tid & 31;
    const int wid  = tid >> 5;      // 0..7 waves
    const int wm   = wid >> 2;      // 0..1 : 64-row slab
    const int wn   = wid & 3;       // 0..3 : 32-col slab
    const int hi   = lane >> 4;
    const int l15  = lane & 15;

    const int bm0 = blockIdx.y * BM;
    const int bn0 = blockIdx.x * BN;

    const bf16*  Ab = (const bf16*)Aptr;
    const float* Af = (const float*)Aptr;

    // staging coordinates for VALU paths: rows r0 and r0+64, 8-element chunk c0
    const int r0 = tid >> 2;
    const int c0 = (tid & 3) * 8;

    v8f acc[4][2];
#pragma unroll
    for (int mi = 0; mi < 4; ++mi)
#pragma unroll
        for (int ni = 0; ni < 2; ++ni)
            acc[mi][ni] = (v8f)0.0f;

    float4 pf[4];   // f32-A prefetch registers (software pipeline)

    auto loadA_f32 = [&](int k0) {
        size_t base = (size_t)bm0 * (size_t)K + (size_t)k0;
        const float* p0 = Af + base + (size_t)r0 * K + c0;
        const float* p1 = Af + base + (size_t)(r0 + 64) * K + c0;
        pf[0] = *(const float4*)p0; pf[1] = *(const float4*)(p0 + 4);
        pf[2] = *(const float4*)p1; pf[3] = *(const float4*)(p1 + 4);
    };
    auto storeA_f32 = [&](int buf) {
        union { uint4 u; bf16 h[8]; } t0, t1;
        t0.h[0] = (bf16)pf[0].x; t0.h[1] = (bf16)pf[0].y;
        t0.h[2] = (bf16)pf[0].z; t0.h[3] = (bf16)pf[0].w;
        t0.h[4] = (bf16)pf[1].x; t0.h[5] = (bf16)pf[1].y;
        t0.h[6] = (bf16)pf[1].z; t0.h[7] = (bf16)pf[1].w;
        t1.h[0] = (bf16)pf[2].x; t1.h[1] = (bf16)pf[2].y;
        t1.h[2] = (bf16)pf[2].z; t1.h[3] = (bf16)pf[2].w;
        t1.h[4] = (bf16)pf[3].x; t1.h[5] = (bf16)pf[3].y;
        t1.h[6] = (bf16)pf[3].z; t1.h[7] = (bf16)pf[3].w;
        *(uint4*)&Ash[buf][r0 * LDS_STRIDE + c0]        = t0.u;
        *(uint4*)&Ash[buf][(r0 + 64) * LDS_STRIDE + c0] = t1.u;
    };
    auto valu_stage_A_bf16 = [&](int k0, int buf) {
        size_t base = (size_t)bm0 * (size_t)K + (size_t)k0;
        *(uint4*)&Ash[buf][r0 * LDS_STRIDE + c0] =
            *(const uint4*)(Ab + base + (size_t)r0 * K + c0);
        *(uint4*)&Ash[buf][(r0 + 64) * LDS_STRIDE + c0] =
            *(const uint4*)(Ab + base + (size_t)(r0 + 64) * K + c0);
    };
    auto valu_stage_B = [&](int k0, int buf) {
        size_t base = (size_t)bn0 * (size_t)K + (size_t)k0;
        *(uint4*)&Bsh[buf][r0 * LDS_STRIDE + c0] =
            *(const uint4*)(Bptr + base + (size_t)r0 * K + c0);
        *(uint4*)&Bsh[buf][(r0 + 64) * LDS_STRIDE + c0] =
            *(const uint4*)(Bptr + base + (size_t)(r0 + 64) * K + c0);
    };

    auto compute_tile = [&](int buf) {
        v16bf afrag[4], bfrag[2];
#pragma unroll
        for (int mi = 0; mi < 4; ++mi) {
            // A: lane holds row (lane&15); VGPR0-3: K=hi*8+0..7, VGPR4-7: +16
            const bf16* p = &Ash[buf][(wm * 64 + mi * 16 + l15) * LDS_STRIDE + hi * 8];
            ((uint4*)&afrag[mi])[0] = *(const uint4*)p;
            ((uint4*)&afrag[mi])[1] = *(const uint4*)(p + 16);
        }
#pragma unroll
        for (int ni = 0; ni < 2; ++ni) {
            // B: lane holds col (lane&15); VGPR0-3: K=hi*8+0..7, VGPR4-7: +16
            const bf16* p = &Bsh[buf][(wn * 32 + ni * 16 + l15) * LDS_STRIDE + hi * 8];
            ((uint4*)&bfrag[ni])[0] = *(const uint4*)p;
            ((uint4*)&bfrag[ni])[1] = *(const uint4*)(p + 16);
        }
#pragma unroll
        for (int mi = 0; mi < 4; ++mi)
#pragma unroll
            for (int ni = 0; ni < 2; ++ni)
                acc[mi][ni] = __builtin_amdgcn_wmma_f32_16x16x32_bf16(
                    false, afrag[mi], false, bfrag[ni],
                    (short)0, acc[mi][ni], false, false);
    };

    const int nT = K / BK;

#if GFX1250_TDM
    constexpr short PEND = A_F32 ? 1 : 2;   // TDM ops issued per tile
    // prologue: stage tile 0 into buffer 0
    if constexpr (A_F32) {
        loadA_f32(0);
        if (wid == 0)
            tdm_load_tile_bf16(Bptr + (size_t)bn0 * K, lds_byte_addr(&Bsh[0][0]), K);
        storeA_f32(0);
    } else {
        if (wid == 0) {
            tdm_load_tile_bf16(Ab + (size_t)bm0 * K, lds_byte_addr(&Ash[0][0]), K);
            tdm_load_tile_bf16(Bptr + (size_t)bn0 * K, lds_byte_addr(&Bsh[0][0]), K);
        }
    }
    for (int t = 0; t < nT; ++t) {
        const int cur = t & 1, nxt = cur ^ 1;
        const int k1  = (t + 1) * BK;
        if (t + 1 < nT) {
            if constexpr (A_F32) loadA_f32(k1);
            if (wid == 0) {
                if constexpr (!A_F32)
                    tdm_load_tile_bf16(Ab + (size_t)bm0 * K + k1,
                                       lds_byte_addr(&Ash[nxt][0]), K);
                tdm_load_tile_bf16(Bptr + (size_t)bn0 * K + k1,
                                   lds_byte_addr(&Bsh[nxt][0]), K);
                // in-order TDM: <= PEND outstanding => tile t fully in LDS
                __builtin_amdgcn_s_wait_tensorcnt(PEND);
            }
        } else {
            if (wid == 0) __builtin_amdgcn_s_wait_tensorcnt(0);
        }
        __syncthreads();
        compute_tile(cur);
        __syncthreads();
        if constexpr (A_F32) { if (t + 1 < nT) storeA_f32(nxt); }
    }
#else
    // fallback: synchronous single-buffer staging through VGPRs
    for (int t = 0; t < nT; ++t) {
        const int k0 = t * BK;
        if constexpr (A_F32) { loadA_f32(k0); storeA_f32(0); }
        else                 { valu_stage_A_bf16(k0, 0); }
        valu_stage_B(k0, 0);
        __syncthreads();
        compute_tile(0);
        __syncthreads();
    }
#endif

    // ---- epilogue: C layout: VGPR r -> M = r + hi*8 ; lane&15 -> N ----
#pragma unroll
    for (int ni = 0; ni < 2; ++ni) {
        const int gn = bn0 + wn * 32 + ni * 16 + l15;
        float bb = 0.0f;
        if constexpr (HAS_BIAS) bb = bias[gn];
#pragma unroll
        for (int mi = 0; mi < 4; ++mi) {
#pragma unroll
            for (int r = 0; r < 8; ++r) {
                const int gm = bm0 + wm * 64 + mi * 16 + hi * 8 + r;
                const float v = acc[mi][ni][r] + bb;
                if constexpr (OUT_F32)
                    Cf[(size_t)gm * (size_t)N + gn] = v;
                if constexpr (OUT_BF16) {
                    if constexpr (OUT_T) Cb[(size_t)gn * (size_t)M + gm] = (bf16)v;
                    else                 Cb[(size_t)gm * (size_t)N + gn] = (bf16)v;
                }
            }
        }
    }
}

// ---------------- elementwise helpers ----------------
__global__ void __launch_bounds__(256)
f32_to_bf16_kernel(const float* __restrict__ src, bf16* __restrict__ dst, int n)
{
    int i = (blockIdx.x * 256 + threadIdx.x) * 4;
    if (i + 3 < n) {
        float4 v = *(const float4*)(src + i);
        union { unsigned long long u; bf16 h[4]; } t;
        t.h[0] = (bf16)v.x; t.h[1] = (bf16)v.y;
        t.h[2] = (bf16)v.z; t.h[3] = (bf16)v.w;
        *(unsigned long long*)(dst + i) = t.u;
    }
}

// W [1024,1024] f32 row-major -> W^T bf16 [1024,1024] (so GEMM B is [N,K])
__global__ void __launch_bounds__(256)
f32_to_bf16_transpose_kernel(const float* __restrict__ src, bf16* __restrict__ dst)
{
    int i = blockIdx.x * 256 + threadIdx.x;     // over 1024*1024
    int k = i >> 10;
    int n = i & (HIDDEN - 1);
    dst[((size_t)n << 10) + k] = (bf16)src[i];
}

// speech = speech_embed[:, ::4] * conv_w  -> bf16
__global__ void __launch_bounds__(256)
speech_subsample_kernel(const float* __restrict__ se, const float* __restrict__ convw,
                        bf16* __restrict__ dst)
{
    int i = blockIdx.x * 256 + threadIdx.x;     // over NSEQ*HIDDEN
    int row = i >> 10;
    int col = i & (HIDDEN - 1);
    float4 v = *(const float4*)(se + (size_t)row * (HIDDEN * 4) + (size_t)col * 4);
    dst[i] = (bf16)(v.x * convw[0]);
}

// row-wise softmax, in place, N = 8192, one block per row, row held in VGPRs
__global__ void __launch_bounds__(256)
softmax_rows_kernel(float* __restrict__ data, int N)
{
    const int t = threadIdx.x;
    float4* rp = (float4*)(data + (size_t)blockIdx.x * (size_t)N);

    float4 v[8];
#pragma unroll
    for (int i = 0; i < 8; ++i) v[i] = rp[i * 256 + t];

    float m = -3.4e38f;
#pragma unroll
    for (int i = 0; i < 8; ++i)
        m = fmaxf(m, fmaxf(fmaxf(v[i].x, v[i].y), fmaxf(v[i].z, v[i].w)));
#pragma unroll
    for (int o = 16; o > 0; o >>= 1) m = fmaxf(m, __shfl_xor(m, o, 32));

    __shared__ float red[8];
    if ((t & 31) == 0) red[t >> 5] = m;
    __syncthreads();
    m = red[0];
#pragma unroll
    for (int i = 1; i < 8; ++i) m = fmaxf(m, red[i]);
    __syncthreads();

    float s = 0.0f;
#pragma unroll
    for (int i = 0; i < 8; ++i) {
        v[i].x = __expf(v[i].x - m);
        v[i].y = __expf(v[i].y - m);
        v[i].z = __expf(v[i].z - m);
        v[i].w = __expf(v[i].w - m);
        s += (v[i].x + v[i].y) + (v[i].z + v[i].w);
    }
#pragma unroll
    for (int o = 16; o > 0; o >>= 1) s += __shfl_xor(s, o, 32);
    if ((t & 31) == 0) red[t >> 5] = s;
    __syncthreads();
    s = 0.0f;
#pragma unroll
    for (int i = 0; i < 8; ++i) s += red[i];

    float inv = 1.0f / s;
#pragma unroll
    for (int i = 0; i < 8; ++i) {
        v[i].x *= inv; v[i].y *= inv; v[i].z *= inv; v[i].w *= inv;
        rp[i * 256 + t] = v[i];
    }
}

// ---------------- driver ----------------
extern "C" void kernel_launch(void* const* d_in, const int* in_sizes, int n_in,
                              void* d_out, int out_size, void* d_ws, size_t ws_size,
                              hipStream_t stream)
{
    (void)in_sizes; (void)n_in; (void)out_size; (void)ws_size;

    const float* speech_embed = (const float*)d_in[0];
    const float* text_embed   = (const float*)d_in[1];
    const float* conv_w       = (const float*)d_in[2];
    const float* Wq = (const float*)d_in[3];  const float* bq = (const float*)d_in[4];
    const float* Wk = (const float*)d_in[5];  const float* bk = (const float*)d_in[6];
    const float* Wv = (const float*)d_in[7];  const float* bv = (const float*)d_in[8];
    const float* Wo = (const float*)d_in[9];  const float* bo = (const float*)d_in[10];

    float* out      = (float*)d_out;
    float* tts_out  = out;
    float* tts_w    = tts_out + (size_t)NSEQ * HIDDEN;
    float* stt_out  = tts_w   + (size_t)NSEQ * NSEQ;
    float* stt_w    = stt_out + (size_t)NSEQ * HIDDEN;

    // workspace carve-out (~104 MB)
    char*  ws  = (char*)d_ws;
    size_t off = 0;
    const size_t NH = (size_t)NSEQ * HIDDEN * sizeof(bf16);
    const size_t HH = (size_t)HIDDEN * HIDDEN * sizeof(bf16);
    bf16* text_b   = (bf16*)(ws + off); off += NH;
    bf16* speech_b = (bf16*)(ws + off); off += NH;
    bf16* WqT_b    = (bf16*)(ws + off); off += HH;
    bf16* WkT_b    = (bf16*)(ws + off); off += HH;
    bf16* WvT_b    = (bf16*)(ws + off); off += HH;
    bf16* WoT_b    = (bf16*)(ws + off); off += HH;
    bf16* q_b      = (bf16*)(ws + off); off += NH;   // [NSEQ, HIDDEN]
    bf16* k_b      = (bf16*)(ws + off); off += NH;   // [NSEQ, HIDDEN]
    bf16* vT_b     = (bf16*)(ws + off); off += NH;   // [HIDDEN, NSEQ] (transposed)
    bf16* attn_b   = (bf16*)(ws + off); off += NH;   // [NSEQ, HIDDEN]

    // ---- stage 0: conversions ----
    {
        int n = NSEQ * HIDDEN;
        f32_to_bf16_kernel<<<n / 4 / 256, 256, 0, stream>>>(text_embed, text_b, n);
        speech_subsample_kernel<<<n / 256, 256, 0, stream>>>(speech_embed, conv_w, speech_b);
        int w = HIDDEN * HIDDEN;
        f32_to_bf16_transpose_kernel<<<w / 256, 256, 0, stream>>>(Wq, WqT_b);
        f32_to_bf16_transpose_kernel<<<w / 256, 256, 0, stream>>>(Wk, WkT_b);
        f32_to_bf16_transpose_kernel<<<w / 256, 256, 0, stream>>>(Wv, WvT_b);
        f32_to_bf16_transpose_kernel<<<w / 256, 256, 0, stream>>>(Wo, WoT_b);
    }

    const bf16* qsrc[2]  = { text_b,   speech_b };
    const bf16* kvsrc[2] = { speech_b, text_b   };
    float*      wmat[2]  = { tts_w,    stt_w    };
    float*      omat[2]  = { tts_out,  stt_out  };

    dim3 gproj(HIDDEN / BN, NSEQ / BM);   // [8192,1024] outputs
    dim3 gsc  (NSEQ   / BN, NSEQ / BM);   // [8192,8192] scores

    for (int p = 0; p < 2; ++p) {
        // Q/K/V projections: x @ W + b    (B = W^T stored [N,K])
        gemm_bf16_wmma<false, true, false, true, false><<<gproj, GEMM_THREADS, 0, stream>>>(
            qsrc[p],  WqT_b, bq, nullptr, q_b, NSEQ, HIDDEN, HIDDEN);
        gemm_bf16_wmma<false, true, false, true, false><<<gproj, GEMM_THREADS, 0, stream>>>(
            kvsrc[p], WkT_b, bk, nullptr, k_b, NSEQ, HIDDEN, HIDDEN);
        // v projection: output stored transposed -> vT_b [HIDDEN, NSEQ]
        gemm_bf16_wmma<false, true, false, true, true><<<gproj, GEMM_THREADS, 0, stream>>>(
            kvsrc[p], WvT_b, bv, nullptr, vT_b, NSEQ, HIDDEN, HIDDEN);

        // S = q @ k^T -> f32 directly into the weights output region (B = k, [N,K])
        gemm_bf16_wmma<false, false, true, false, false><<<gsc, GEMM_THREADS, 0, stream>>>(
            q_b, k_b, nullptr, wmat[p], nullptr, NSEQ, NSEQ, HIDDEN);

        // softmax rows in place (this region IS the reference's `weights` output)
        softmax_rows_kernel<<<NSEQ, 256, 0, stream>>>(wmat[p], NSEQ);

        // attn = weights @ v   (A f32 from d_out, B = vT_b [N,K])
        gemm_bf16_wmma<true, false, false, true, false><<<gproj, GEMM_THREADS, 0, stream>>>(
            wmat[p], vT_b, nullptr, nullptr, attn_b, NSEQ, HIDDEN, NSEQ);

        // out = attn @ Wo + bo -> f32 output  (B = Wo^T [N,K])
        gemm_bf16_wmma<false, true, true, false, false><<<gproj, GEMM_THREADS, 0, stream>>>(
            attn_b, WoT_b, bo, omat[p], nullptr, NSEQ, HIDDEN, HIDDEN);
    }
}
